// LSTMModel_14826227106162
// MI455X (gfx1250) — matrix operand order repaired and checked by reading, not compiled, not run on hardware
//
#include <hip/hip_runtime.h>
#include <hip/hip_bf16.h>
#include <math.h>

typedef __attribute__((ext_vector_type(16))) _Float16 v16h;
typedef __attribute__((ext_vector_type(8)))  _Float16 v8h;
typedef __attribute__((ext_vector_type(8)))  float    v8f;
typedef __attribute__((ext_vector_type(4)))  unsigned int v4u;
typedef __attribute__((ext_vector_type(8)))  int          v8i;
typedef __attribute__((ext_vector_type(4)))  int          v4i;

#define B_      32
#define T_      64
#define IN_DIM_ 2048
#define E_      512
#define H_      1024
#define V_      10000
#define NL_     2
#define G4H_    (4 * H_)

// Tensor Data Mover availability (arity differs between toolchains).
#if defined(__has_builtin)
#if __has_builtin(__builtin_amdgcn_tensor_load_to_lds) && \
    __has_builtin(__builtin_amdgcn_s_wait_tensorcnt)
#define USE_TDM 1
#endif
#endif
#ifndef USE_TDM
#define USE_TDM 0
#endif

// ---------------------------------------------------------------------------
// WMMA fragment loaders (CDNA5 wave32, V_WMMA_F32_16X16X32_F16 layouts).
// A: 16x32 f16 row-major. Lane L (0-15): M=L, halves 0-7 = K k0..k0+7,
//    halves 8-15 = K k0+16..23. Lane L+16: M=L, K k0+8..15 / k0+24..31.
// B: 32x16 f16, weight stored [N,K] row-major (B = W^T). Lane n (0-15) holds
//    column n, K = k0..k0+7 / k0+16..23; lanes 16-31 hold K = k0+8..15 / k0+24..31.
// The A loader is used on LDS slabs (emits ds_load_b128).
// ---------------------------------------------------------------------------
static __device__ __forceinline__ v16h load_a_frag(const _Float16* A, int lda,
                                                   int m0, int k0, int lane) {
    const int m  = m0 + (lane & 15);
    const int kA = k0 + ((lane & 16) ? 8 : 0);
    const _Float16* p = A + (size_t)m * lda;
    v8h lo = *(const v8h*)(p + kA);
    v8h hi = *(const v8h*)(p + kA + 16);
    v16h r;
#pragma unroll
    for (int i = 0; i < 8; ++i) { r[i] = lo[i]; r[8 + i] = hi[i]; }
    return r;
}

static __device__ __forceinline__ v16h load_b_frag(const _Float16* __restrict__ W,
                                                   int ldw, int n0, int k0, int lane) {
    const int n = n0 + (lane & 15);
    const int k = k0 + ((lane & 16) ? 8 : 0);
    const _Float16* p = W + (size_t)n * ldw + k;
    v8h lo = *(const v8h*)(p);
    v8h hi = *(const v8h*)(p + 16);
    v16h r;
#pragma unroll
    for (int i = 0; i < 8; ++i) { r[i] = lo[i]; r[8 + i] = hi[i]; }
    return r;
}

#if USE_TDM
// ---------------------------------------------------------------------------
// TDM: DMA a contiguous 32 x Khalfs f16 slab (lda == K) from global into LDS
// at byte offset ldsOff. 2D descriptor per ISA D# (CDNA5 ISA):
//   tile_dim0 = K (2B units), tile_dim1 = 32, tensor_dim0_stride = K,
//   data_size = 1 (2 bytes), type = 2 ("image"), count = 1.
// Issued by one wave; completion via s_wait_tensorcnt, others join at barrier.
// (No static LDS in this kernel, so dynamic-shared slab begins at LDS addr 0.)
// ---------------------------------------------------------------------------
static __device__ __forceinline__ void tdm_load_slab(const _Float16* gptr,
                                                     unsigned ldsOff, int Khalfs) {
    const unsigned long long ga = (unsigned long long)(size_t)gptr;
    v4u g0;
    g0[0] = 1u;                                          // count=1, user mode
    g0[1] = ldsOff;                                      // lds_addr (bytes)
    g0[2] = (unsigned)(ga & 0xffffffffu);                // global_addr[31:0]
    g0[3] = (unsigned)((ga >> 32) & 0x1ffffffu) | (2u << 30);  // addr[56:32]|type=2
    v8i g1;
    g1[0] = 0x00010000;                                  // data_size=2B, no opts
    g1[1] = (Khalfs & 0xffff) << 16;                     // tensor_dim0[15:0]
    g1[2] = ((Khalfs >> 16) & 0xffff) | (32 << 16);      // td0[31:16] | tensor_dim1=32
    g1[3] = (Khalfs & 0xffff) << 16;                     // tile_dim0 = Khalfs
    g1[4] = 32;                                          // tile_dim1 = 32
    g1[5] = Khalfs;                                      // tensor_dim0_stride[31:0]
    g1[6] = 0;                                           // stride hi | dim1_stride lo
    g1[7] = 0;
    v4i z4 = {0, 0, 0, 0};
#if __clang_major__ >= 23
    v8i z8 = {0, 0, 0, 0, 0, 0, 0, 0};
    __builtin_amdgcn_tensor_load_to_lds(g0, g1, z4, z4, z8, 0);
#else
    __builtin_amdgcn_tensor_load_to_lds(g0, g1, z4, z4, 0);
#endif
}
#endif

// ---------------------------------------------------------------------------
// Register-blocked GEMM with LDS-staged A slab (TDM DMA when available).
//   C[M,N] = A1[M,K1] @ W1[N,K1]^T (+ A2 @ W2^T if DUAL) + bias1 + bias2
// Block = 256 threads (8 waves). Each block owns a 32-row macro-row
// (blockIdx.y); its A slab(s) are staged into LDS once (lda==K: contiguous).
// Each wave computes a 32 x (NT*16) tile: 2 A-frags (from LDS) reused across
// NT B-frags -> 2*NT WMMAs per fragment-load group.
// ---------------------------------------------------------------------------
template <int NT, bool DUAL>
__global__ void __launch_bounds__(256)
gemm_wmma_blk(const _Float16* __restrict__ A1, int lda1, int K1,
              const _Float16* __restrict__ W1,
              const _Float16* __restrict__ A2, int lda2, int K2,
              const _Float16* __restrict__ W2,
              const float* __restrict__ bias1, const float* __restrict__ bias2,
              float* __restrict__ C, int ldc, int N) {
    extern __shared__ _Float16 smem[];
    _Float16* As1 = smem;                 // 32 x K1
    _Float16* As2 = smem + 32 * K1;       // 32 x K2 (DUAL only)

    const int lane = threadIdx.x & 31;
    const int tid  = threadIdx.x;

    // macro-row offset
    A1 += (size_t)blockIdx.y * 32 * lda1;
    if (DUAL) A2 += (size_t)blockIdx.y * 32 * lda2;
    C  += (size_t)blockIdx.y * 32 * ldc;

#if USE_TDM
    if (tid < 32) {                                       // wave 0 drives the DMA
        tdm_load_slab(A1, 0u, K1);
        if (DUAL) tdm_load_slab(A2, (unsigned)(32 * K1 * 2), K2);
        __builtin_amdgcn_s_wait_tensorcnt(0);
    }
#else
    for (int c = tid; c < 4 * K1; c += 256) {             // 32*K1/8 chunks
        const int m = c / (K1 >> 3), k = (c % (K1 >> 3)) << 3;
        *(v8h*)(As1 + m * K1 + k) = *(const v8h*)(A1 + (size_t)m * lda1 + k);
    }
    if (DUAL) {
        for (int c = tid; c < 4 * K2; c += 256) {
            const int m = c / (K2 >> 3), k = (c % (K2 >> 3)) << 3;
            *(v8h*)(As2 + m * K2 + k) = *(const v8h*)(A2 + (size_t)m * lda2 + k);
        }
    }
#endif
    __syncthreads();

    const int wcol = blockIdx.x * 8 + (tid >> 5);         // wave's column group
    const int n0 = wcol * (NT * 16);
    if (n0 >= N) return;                                  // whole-wave exit

    v8f acc[2][NT];
#pragma unroll
    for (int i = 0; i < 2; ++i)
#pragma unroll
        for (int j = 0; j < NT; ++j) acc[i][j] = (v8f){};

    for (int k0 = 0; k0 < K1; k0 += 32) {
        const v16h a0 = load_a_frag(As1, K1, 0,  k0, lane);   // ds_load
        const v16h a1 = load_a_frag(As1, K1, 16, k0, lane);
        if (k0 + 32 < K1)
            __builtin_prefetch(W1 + (size_t)(n0 + (lane & 15)) * K1 + k0 + 32, 0, 3);
#pragma unroll
        for (int nt = 0; nt < NT; ++nt) {
            const v16h b = load_b_frag(W1, K1, n0 + nt * 16, k0, lane);
            acc[0][nt] = __builtin_amdgcn_wmma_f32_16x16x32_f16(
                false, a0, false, b, (short)0, acc[0][nt], false, false);
            acc[1][nt] = __builtin_amdgcn_wmma_f32_16x16x32_f16(
                false, a1, false, b, (short)0, acc[1][nt], false, false);
        }
    }
    if (DUAL) {
        for (int k0 = 0; k0 < K2; k0 += 32) {
            const v16h a0 = load_a_frag(As2, K2, 0,  k0, lane);
            const v16h a1 = load_a_frag(As2, K2, 16, k0, lane);
            if (k0 + 32 < K2)
                __builtin_prefetch(W2 + (size_t)(n0 + (lane & 15)) * K2 + k0 + 32, 0, 3);
#pragma unroll
            for (int nt = 0; nt < NT; ++nt) {
                const v16h b = load_b_frag(W2, K2, n0 + nt * 16, k0, lane);
                acc[0][nt] = __builtin_amdgcn_wmma_f32_16x16x32_f16(
                    false, a0, false, b, (short)0, acc[0][nt], false, false);
                acc[1][nt] = __builtin_amdgcn_wmma_f32_16x16x32_f16(
                    false, a1, false, b, (short)0, acc[1][nt], false, false);
            }
        }
    }

    // store: C/D layout -> VGPR r holds row (mt*16 + (lane&16?8:0) + r), col lane&15
    const int nn = lane & 15;
    const int rb = (lane & 16) ? 8 : 0;
#pragma unroll
    for (int nt = 0; nt < NT; ++nt) {
        const int n = n0 + nt * 16 + nn;
        float bs = 0.0f;
        if (bias1) bs += bias1[n];
        if (bias2) bs += bias2[n];
#pragma unroll
        for (int mt = 0; mt < 2; ++mt)
#pragma unroll
            for (int r = 0; r < 8; ++r)
                C[(size_t)(mt * 16 + rb + r) * ldc + n] = acc[mt][nt][r] + bs;
    }
}

// ---------------------------------------------------------------------------
// Elementwise / utility kernels
// ---------------------------------------------------------------------------
__global__ void f32_to_f16(const float* __restrict__ in, _Float16* __restrict__ out,
                           size_t n) {
    size_t i = (size_t)blockIdx.x * blockDim.x + threadIdx.x;
    const size_t stride = (size_t)gridDim.x * blockDim.x;
    for (; i < n; i += stride) out[i] = (_Float16)in[i];
}

__global__ void embed_gather(const float* __restrict__ emb, const int* __restrict__ y,
                             _Float16* __restrict__ e) {
    size_t i = (size_t)blockIdx.x * blockDim.x + threadIdx.x;
    const size_t n = (size_t)T_ * B_ * E_;
    const size_t stride = (size_t)gridDim.x * blockDim.x;
    for (; i < n; i += stride) {
        const int j = (int)(i % E_);
        const size_t tb = i / E_;
        const int b = (int)(tb % B_);
        const int t = (int)(tb / B_);
        const int row = y[b * T_ + t];
        e[i] = (_Float16)emb[(size_t)row * E_ + j];
    }
}

__global__ void init_state(const float* __restrict__ h0, const float* __restrict__ c0,
                           float* __restrict__ h_st, float* __restrict__ c_st,
                           _Float16* __restrict__ h_st_h) {
    const int n = B_ * H_;
    const int i = blockIdx.x * blockDim.x + threadIdx.x;
    if (i >= n) return;
    const float h = h0[i], c = c0[i];
    h_st[i] = h;      h_st[n + i] = h;
    c_st[i] = c;      c_st[n + i] = c;
    h_st_h[i] = (_Float16)h;  h_st_h[n + i] = (_Float16)h;
}

static __device__ __forceinline__ float sigm(float x) {
    return 1.0f / (1.0f + __expf(-x));
}

__global__ void lstm_cell(const float* __restrict__ g, const float* __restrict__ c0,
                          float* __restrict__ c_st, float* __restrict__ h_st,
                          _Float16* __restrict__ h_st_h,
                          _Float16* __restrict__ out_h, int t) {
    const int idx = blockIdx.x * blockDim.x + threadIdx.x;
    if (idx >= B_ * H_) return;
    const int b = idx >> 10;
    const int j = idx & (H_ - 1);
    const float* gb = g + (size_t)b * G4H_;
    const float i_g = sigm(gb[j]);
    const float f_g = sigm(gb[H_ + j]);
    const float g_g = tanhf(gb[2 * H_ + j]);
    const float o_g = sigm(gb[3 * H_ + j]);
    const float c_raw = f_g * c_st[idx] + i_g * g_g;  // pre-mix cell
    const float h = o_g * tanhf(c_raw);               // h uses RAW cell (per ref)
    c_st[idx] = 0.5f * (c0[idx] + c_raw);             // carried (and final) c
    h_st[idx] = h;
    h_st_h[idx] = (_Float16)h;
    if (out_h) out_h[((size_t)b * T_ + t) * H_ + j] = (_Float16)h;
}

__global__ void softmax_time(float* __restrict__ logits) {
    const int tid = blockIdx.x * blockDim.x + threadIdx.x;
    if (tid >= B_ * V_) return;
    const int b = tid / V_;
    const int v = tid - b * V_;
    float* p = logits + (size_t)b * T_ * V_ + v;
    float mx = -INFINITY;
    for (int t = 0; t < T_; ++t) mx = fmaxf(mx, p[(size_t)t * V_]);
    float s = 0.0f;
    for (int t = 0; t < T_; ++t) s += __expf(p[(size_t)t * V_] - mx);
    const float inv = 1.0f / s;
    for (int t = 0; t < T_; ++t) p[(size_t)t * V_] = __expf(p[(size_t)t * V_] - mx) * inv;
}

__global__ void copy_states(const float* __restrict__ h_st, const float* __restrict__ c_st,
                            float* __restrict__ out_tail) {
    const int n = NL_ * B_ * H_;
    const int i = blockIdx.x * blockDim.x + threadIdx.x;
    if (i >= n) return;
    out_tail[i]     = h_st[i];
    out_tail[n + i] = c_st[i];
}

// ---------------------------------------------------------------------------
// Host orchestration
// ---------------------------------------------------------------------------
extern "C" void kernel_launch(void* const* d_in, const int* in_sizes, int n_in,
                              void* d_out, int out_size, void* d_ws, size_t ws_size,
                              hipStream_t stream) {
    const float* x    = (const float*)d_in[0];
    const int*   y    = (const int*)  d_in[1];
    const float* W_h  = (const float*)d_in[2];
    const float* b_h  = (const float*)d_in[3];
    const float* W_c  = (const float*)d_in[4];
    const float* b_c  = (const float*)d_in[5];
    const float* emb  = (const float*)d_in[6];
    const float* Wih0 = (const float*)d_in[7];
    const float* Whh0 = (const float*)d_in[8];
    const float* bih0 = (const float*)d_in[9];
    const float* bhh0 = (const float*)d_in[10];
    const float* Wih1 = (const float*)d_in[11];
    const float* Whh1 = (const float*)d_in[12];
    const float* bih1 = (const float*)d_in[13];
    const float* bhh1 = (const float*)d_in[14];
    const float* LM_W = (const float*)d_in[15];
    const float* LM_b = (const float*)d_in[16];
    float* out = (float*)d_out;

    // allow >64KB dynamic LDS (CDNA5 WGP has 320KB); harmless if redundant
    (void)hipFuncSetAttribute((const void*)gemm_wmma_blk<2, true>,
                              hipFuncAttributeMaxDynamicSharedMemorySize, 160 * 1024);
    (void)hipFuncSetAttribute((const void*)gemm_wmma_blk<2, false>,
                              hipFuncAttributeMaxDynamicSharedMemorySize, 160 * 1024);
    (void)hipFuncSetAttribute((const void*)gemm_wmma_blk<5, false>,
                              hipFuncAttributeMaxDynamicSharedMemorySize, 160 * 1024);

    // Workspace carve-out (~66 MB): f16 weight/activation copies + fp32 state.
    char* ws = (char*)d_ws;
    size_t off = 0;
    auto alloc_h = [&](size_t n) { _Float16* p = (_Float16*)(ws + off);
                                   off += ((n * 2 + 255) & ~(size_t)255); return p; };
    auto alloc_f = [&](size_t n) { float* p = (float*)(ws + off);
                                   off += ((n * 4 + 255) & ~(size_t)255); return p; };

    _Float16* Wh_h   = alloc_h((size_t)H_ * IN_DIM_);
    _Float16* Wc_h   = alloc_h((size_t)H_ * IN_DIM_);
    _Float16* Wih0_h = alloc_h((size_t)G4H_ * E_);
    _Float16* Whh0_h = alloc_h((size_t)G4H_ * H_);
    _Float16* Wih1_h = alloc_h((size_t)G4H_ * H_);
    _Float16* Whh1_h = alloc_h((size_t)G4H_ * H_);
    _Float16* LMW_h  = alloc_h((size_t)V_ * H_);
    _Float16* x_h    = alloc_h((size_t)B_ * IN_DIM_);
    _Float16* e_h    = alloc_h((size_t)T_ * B_ * E_);
    _Float16* hst_h  = alloc_h((size_t)NL_ * B_ * H_);
    _Float16* out_h  = alloc_h((size_t)B_ * T_ * H_);
    float* h0    = alloc_f((size_t)B_ * H_);
    float* c0    = alloc_f((size_t)B_ * H_);
    float* hst   = alloc_f((size_t)NL_ * B_ * H_);
    float* cst   = alloc_f((size_t)NL_ * B_ * H_);
    float* gates = alloc_f((size_t)B_ * G4H_);

    // 1) fp32 -> f16 conversions (one HBM pass; f16 copies stay L2-resident)
    auto cvt = [&](const float* src, _Float16* dst, size_t n) {
        int blocks = (int)((n + 255) / 256); if (blocks > 2048) blocks = 2048;
        f32_to_f16<<<blocks, 256, 0, stream>>>(src, dst, n);
    };
    cvt(W_h,  Wh_h,   (size_t)H_ * IN_DIM_);
    cvt(W_c,  Wc_h,   (size_t)H_ * IN_DIM_);
    cvt(Wih0, Wih0_h, (size_t)G4H_ * E_);
    cvt(Whh0, Whh0_h, (size_t)G4H_ * H_);
    cvt(Wih1, Wih1_h, (size_t)G4H_ * H_);
    cvt(Whh1, Whh1_h, (size_t)G4H_ * H_);
    cvt(LM_W, LMW_h,  (size_t)V_ * H_);
    cvt(x,    x_h,    (size_t)B_ * IN_DIM_);
    embed_gather<<<2048, 256, 0, stream>>>(emb, y, e_h);

    // 2) init GEMMs: h0/c0 = x @ W^T + b  (M=32, N=1024, K=2048; LDS slab 128KB)
    {
        const dim3 grid(H_ / 32 / 8, 1);                       // 32 wave-cols
        const size_t shm = (size_t)32 * IN_DIM_ * 2;
        gemm_wmma_blk<2, false><<<grid, 256, shm, stream>>>(
            x_h, IN_DIM_, IN_DIM_, Wh_h, nullptr, 0, 0, nullptr,
            b_h, nullptr, h0, H_, H_);
        gemm_wmma_blk<2, false><<<grid, 256, shm, stream>>>(
            x_h, IN_DIM_, IN_DIM_, Wc_h, nullptr, 0, 0, nullptr,
            b_c, nullptr, c0, H_, H_);
        init_state<<<(B_ * H_ + 255) / 256, 256, 0, stream>>>(h0, c0, hst, cst, hst_h);
    }

    // 3) sequential scan: 64 steps x 2 layers; gate GEMM fuses x@Wih^T + h@Whh^T
    const dim3 ggrid(G4H_ / 32 / 8, 1);                        // 128 wave-cols, 16 blocks
    const size_t shm_l0 = (size_t)(32 * E_ + 32 * H_) * 2;     // 96 KB
    const size_t shm_l1 = (size_t)(32 * H_ + 32 * H_) * 2;     // 128 KB (CDNA5-only)
    const int cell_blocks = (B_ * H_ + 255) / 256;
    const size_t BH = (size_t)B_ * H_;
    for (int t = 0; t < T_; ++t) {
        const _Float16* e_t = e_h + (size_t)t * B_ * E_;
        gemm_wmma_blk<2, true><<<ggrid, 256, shm_l0, stream>>>(
            e_t, E_, E_, Wih0_h, hst_h, H_, H_, Whh0_h,
            bih0, bhh0, gates, G4H_, G4H_);
        lstm_cell<<<cell_blocks, 256, 0, stream>>>(gates, c0, cst, hst, hst_h,
                                                   nullptr, t);
        gemm_wmma_blk<2, true><<<ggrid, 256, shm_l1, stream>>>(
            hst_h, H_, H_, Wih1_h, hst_h + BH, H_, H_, Whh1_h,
            bih1, bhh1, gates, G4H_, G4H_);
        lstm_cell<<<cell_blocks, 256, 0, stream>>>(gates, c0, cst + BH, hst + BH,
                                                   hst_h + BH, out_h, t);
    }

    // 4) LM head: logits[2048,10000] = out_h @ LM_W^T + LM_b -> d_out.
    //    32x80 wave tiles (10000 = 125*80), 64 macro-rows, A slab 64KB in LDS.
    {
        const dim3 grid((125 + 7) / 8, (B_ * T_) / 32);        // 16 x 64 blocks
        const size_t shm = (size_t)32 * H_ * 2;
        gemm_wmma_blk<5, false><<<grid, 256, shm, stream>>>(
            out_h, H_, H_, LMW_h, nullptr, 0, 0, nullptr,
            LM_b, nullptr, out, V_, V_);
    }

    // 5) softmax over TIME axis (in place), then append h_f / c_f
    softmax_time<<<(B_ * V_ + 255) / 256, 256, 0, stream>>>(out);
    copy_states<<<(NL_ * B_ * H_ + 255) / 256, 256, 0, stream>>>(
        hst, cst, out + (size_t)B_ * T_ * V_);
}
